// SimpleWorldModel_51393578664645
// MI455X (gfx1250) — compile-verified, matrix operand order checked by reading, not run on hardware
//
#include <hip/hip_runtime.h>
#include <hip/hip_bf16.h>
#include <math.h>

typedef __attribute__((ext_vector_type(16))) _Float16 v16h;
typedef __attribute__((ext_vector_type(8)))  _Float16 v8h;
typedef __attribute__((ext_vector_type(8)))  float    v8f;

#define SIM_DT   (1.0f/60.0f)
#define NBODY    20
#define BATCH    32768
#define KPAD1    416   // 414 padded to multiple of 32
#define HID      512
#define N3VALID  120
#define N3PAD    256

// ---------------------------------------------------------------------------
// Load a 16x32 fp16 WMMA fragment (A or B) from a K-major matrix:
//   base[rc][k], leading dim ld.  Per CDNA5 ISA 7.12.2 (16-bit A 16x32):
//   lane 0-15 : row = lane,     halves 0..7 = K k0..k0+7,    8..15 = K k0+16..k0+23
//   lane16-31 : row = lane-16,  halves 0..7 = K k0+8..k0+15, 8..15 = K k0+24..k0+31
// Both runs are 8 contiguous halfs -> two 16B loads (alignment guaranteed:
// ld is a multiple of 8 halfs, k offsets multiples of 8 halfs).
// ---------------------------------------------------------------------------
__device__ __forceinline__ v16h load_frag16(const _Float16* __restrict__ base,
                                            int ld, int rcbase, int k0, int lane) {
  const int rc = rcbase + (lane & 15);
  const int kg = (lane >> 4) << 3;          // 0 or 8
  const _Float16* p = base + (size_t)rc * ld + k0 + kg;
  v8h lo = *reinterpret_cast<const v8h*>(p);
  v8h hi = *reinterpret_cast<const v8h*>(p + 16);
  v16h f;
#pragma unroll
  for (int i = 0; i < 8; ++i) { f[i] = lo[i]; f[i + 8] = hi[i]; }
  return f;
}

// ---------------------------------------------------------------------------
// Transpose + convert f32 W(K,N) row-major -> fp16 Wt(Npad,Kpad) K-major,
// zero padding out-of-range entries.
// ---------------------------------------------------------------------------
__global__ void conv_transpose(const float* __restrict__ W, _Float16* __restrict__ Wt,
                               int K, int N, int Kpad, int Npad) {
  int idx = blockIdx.x * blockDim.x + threadIdx.x;
  int total = Kpad * Npad;
  if (idx >= total) return;
  int n = idx / Kpad;
  int k = idx - n * Kpad;
  float v = (n < N && k < K) ? W[(size_t)k * N + n] : 0.0f;
  Wt[idx] = (_Float16)v;
}

// ---------------------------------------------------------------------------
// Build X = concat(n_observation, vec6d(flip(quat_from_rotvec(action)))) in fp16,
// padded to KPAD1 columns.  One block per sample.
// ---------------------------------------------------------------------------
__global__ void build_x(const float* __restrict__ obs, const float* __restrict__ act,
                        _Float16* __restrict__ X) {
  const int b = blockIdx.x;
  const int tid = threadIdx.x;
  _Float16* xr = X + (size_t)b * KPAD1;
  const float* ob = obs + (size_t)b * 300;
  for (int c = tid; c < KPAD1; c += blockDim.x) {
    if (c < 300)       xr[c] = (_Float16)ob[c];
    else if (c >= 414) xr[c] = (_Float16)0.0f;   // cols [300,414) filled below
  }
  if (tid < 19) {
    const float* av = act + (size_t)b * 57 + tid * 3;
    float x = av[0], y = av[1], z = av[2];
    float a2 = x * x + y * y + z * z;
    float ang = sqrtf(a2);
    float half = 0.5f * ang;
    float s = (ang < 1e-8f) ? (0.5f - a2 * (1.0f / 48.0f)) : (sinf(half) / ang);
    float qx = x * s, qy = y * s, qz = z * s, qw = cosf(half);
    if (qw < 0.0f) { qx = -qx; qy = -qy; qz = -qz; qw = -qw; }
    float r00 = 1.0f - 2.0f * (qy * qy + qz * qz);
    float r01 = 2.0f * (qx * qy - qz * qw);
    float r10 = 2.0f * (qx * qy + qz * qw);
    float r11 = 1.0f - 2.0f * (qx * qx + qz * qz);
    float r20 = 2.0f * (qx * qz - qy * qw);
    float r21 = 2.0f * (qy * qz + qx * qw);
    _Float16* o = xr + 300 + tid * 6;
    o[0] = (_Float16)r00; o[1] = (_Float16)r01; o[2] = (_Float16)r10;
    o[3] = (_Float16)r11; o[4] = (_Float16)r20; o[5] = (_Float16)r21;
  }
}

// ---------------------------------------------------------------------------
// GEMM (relu, fp16 out): C(M,N) = relu(A(M,K) @ Bt(N,K)^T + bias).
// Block = 256 threads = 8 waves; block tile 32(M) x 256(N); each wave does a
// 16x64 strip (4 accumulators sharing one A fragment per K-step).
// ---------------------------------------------------------------------------
__global__ void __launch_bounds__(256)
gemm_relu_f16(const _Float16* __restrict__ A, const _Float16* __restrict__ Bt,
              const float* __restrict__ bias, _Float16* __restrict__ O,
              int K, int lda, int ldbt, int ldo) {
  const int tid = threadIdx.x;
  const int lane = tid & 31;
  const int wid  = tid >> 5;
  const int mbase = blockIdx.y * 32 + (wid >> 2) * 16;
  const int nbase = blockIdx.x * 256 + (wid & 3) * 64;

  v8f acc[4] = {};
  for (int k0 = 0; k0 < K; k0 += 32) {
    v16h a = load_frag16(A, lda, mbase, k0, lane);
#pragma unroll
    for (int s = 0; s < 4; ++s) {
      v16h bfrag = load_frag16(Bt, ldbt, nbase + s * 16, k0, lane);
      acc[s] = __builtin_amdgcn_wmma_f32_16x16x32_f16(
          false, a, false, bfrag, (short)0, acc[s], false, false);
    }
  }
  // C/D layout: VGPR r, lane 0-15 -> (M=r,N=lane); lane 16-31 -> (M=r+8,N=lane-16)
  const int n0   = lane & 15;
  const int mrow = mbase + ((lane >> 4) << 3);
#pragma unroll
  for (int s = 0; s < 4; ++s) {
    const int n = nbase + s * 16 + n0;
    const float bv = bias[n];
#pragma unroll
    for (int r = 0; r < 8; ++r) {
      float v = fmaxf(acc[s][r] + bv, 0.0f);
      O[(size_t)(mrow + r) * ldo + n] = (_Float16)v;
    }
  }
}

// ---------------------------------------------------------------------------
// Final GEMM: delta(M,120) = (A @ W3 + b3) * delta_std + delta_mean (f32 out).
// Bt is padded to N3PAD rows (zeros) so fragment loads never go OOB.
// ---------------------------------------------------------------------------
__global__ void __launch_bounds__(256)
gemm_delta(const _Float16* __restrict__ A, const _Float16* __restrict__ Bt,
           const float* __restrict__ bias, const float* __restrict__ dstd,
           const float* __restrict__ dmean, float* __restrict__ O, int K) {
  const int tid = threadIdx.x;
  const int lane = tid & 31;
  const int wid  = tid >> 5;
  const int mbase = blockIdx.y * 32 + (wid >> 2) * 16;
  const int nbase = (wid & 3) * 64;   // gridDim.x == 1

  v8f acc[4] = {};
  for (int k0 = 0; k0 < K; k0 += 32) {
    v16h a = load_frag16(A, HID, mbase, k0, lane);
#pragma unroll
    for (int s = 0; s < 4; ++s) {
      v16h bfrag = load_frag16(Bt, HID, nbase + s * 16, k0, lane);
      acc[s] = __builtin_amdgcn_wmma_f32_16x16x32_f16(
          false, a, false, bfrag, (short)0, acc[s], false, false);
    }
  }
  const int n0   = lane & 15;
  const int mrow = mbase + ((lane >> 4) << 3);
#pragma unroll
  for (int s = 0; s < 4; ++s) {
    const int n = nbase + s * 16 + n0;
    if (n < N3VALID) {
      const float bv = bias[n], sd = dstd[n], mn = dmean[n];
#pragma unroll
      for (int r = 0; r < 8; ++r) {
        float v = acc[s][r] + bv;
        O[(size_t)(mrow + r) * N3VALID + n] = v * sd + mn;
      }
    }
  }
}

// ---------------------------------------------------------------------------
// integrate_state: one thread per (batch, body).
// ---------------------------------------------------------------------------
__global__ void integrate(const float* __restrict__ state, const float* __restrict__ delta,
                          float* __restrict__ out) {
  int idx = blockIdx.x * blockDim.x + threadIdx.x;
  if (idx >= BATCH * NBODY) return;
  const int b = idx / NBODY;
  const int j = idx - b * NBODY;
  const float* st = state + (size_t)idx * 13;
  const float* rq = state + (size_t)b * NBODY * 13 + 3;   // root (body 0) quat
  const float rx = rq[0], ry = rq[1], rz = rq[2], rw = rq[3];
  const float* d = delta + (size_t)b * N3VALID + j * 6;

  // rotate lin (d[0..2]) and ang (d[3..5]) by root quat: v + w*t + cross(xyz,t), t = 2*cross(xyz,v)
  float res[6];
#pragma unroll
  for (int h = 0; h < 2; ++h) {
    float vx = d[h * 3 + 0], vy = d[h * 3 + 1], vz = d[h * 3 + 2];
    float tx = 2.0f * (ry * vz - rz * vy);
    float ty = 2.0f * (rz * vx - rx * vz);
    float tz = 2.0f * (rx * vy - ry * vx);
    res[h * 3 + 0] = vx + rw * tx + (ry * tz - rz * ty);
    res[h * 3 + 1] = vy + rw * ty + (rz * tx - rx * tz);
    res[h * 3 + 2] = vz + rw * tz + (rx * ty - ry * tx);
  }
  float px = st[0], py = st[1], pz = st[2];
  float qx = st[3], qy = st[4], qz = st[5], qw = st[6];
  float vx = st[7] + res[0], vy = st[8] + res[1], vz = st[9] + res[2];
  float ax = st[10] + res[3], ay = st[11] + res[4], az = st[12] + res[5];
  px += vx * SIM_DT; py += vy * SIM_DT; pz += vz * SIM_DT;
  // dq = quat_multiply((ax,ay,az,0), q)
  float dx = qw * ax + (ay * qz - az * qy);
  float dy = qw * ay + (az * qx - ax * qz);
  float dz = qw * az + (ax * qy - ay * qx);
  float dw = -(ax * qx + ay * qy + az * qz);
  const float hdt = 0.5f * SIM_DT;
  float nx = qx + hdt * dx, ny = qy + hdt * dy, nz = qz + hdt * dz, nw = qw + hdt * dw;
  float inv = rsqrtf(nx * nx + ny * ny + nz * nz + nw * nw);
  float* o = out + (size_t)idx * 13;
  o[0] = px;  o[1] = py;  o[2] = pz;
  o[3] = nx * inv; o[4] = ny * inv; o[5] = nz * inv; o[6] = nw * inv;
  o[7] = vx;  o[8] = vy;  o[9] = vz;
  o[10] = ax; o[11] = ay; o[12] = az;
}

// ---------------------------------------------------------------------------
// Host launcher.  Inputs (all f32, setup_inputs order):
//  0:state 1:action 2:n_observation 3:W1 4:b1 5:W2 6:b2 7:W3 8:b3 9:delta_mean 10:delta_std
// ---------------------------------------------------------------------------
extern "C" void kernel_launch(void* const* d_in, const int* in_sizes, int n_in,
                              void* d_out, int out_size, void* d_ws, size_t ws_size,
                              hipStream_t stream) {
  const float* state = (const float*)d_in[0];
  const float* action = (const float*)d_in[1];
  const float* obs   = (const float*)d_in[2];
  const float* W1    = (const float*)d_in[3];
  const float* b1    = (const float*)d_in[4];
  const float* W2    = (const float*)d_in[5];
  const float* b2    = (const float*)d_in[6];
  const float* W3    = (const float*)d_in[7];
  const float* b3    = (const float*)d_in[8];
  const float* dmean = (const float*)d_in[9];
  const float* dstd  = (const float*)d_in[10];

  char* ws = (char*)d_ws;
  const size_t SZ_XH   = (size_t)BATCH * HID * sizeof(_Float16);      // 33.55 MB (X uses ld 416; reused as H2)
  const size_t SZ_H1   = (size_t)BATCH * HID * sizeof(_Float16);      // 33.55 MB
  const size_t SZ_W1T  = (size_t)HID * KPAD1 * sizeof(_Float16);
  const size_t SZ_W2T  = (size_t)HID * HID * sizeof(_Float16);
  const size_t SZ_W3T  = (size_t)N3PAD * HID * sizeof(_Float16);
  size_t off = 0;
  _Float16* X   = (_Float16*)(ws + off); off += SZ_XH;    // later reused as H2
  _Float16* H1  = (_Float16*)(ws + off); off += SZ_H1;
  _Float16* W1t = (_Float16*)(ws + off); off += SZ_W1T;
  _Float16* W2t = (_Float16*)(ws + off); off += SZ_W2T;
  _Float16* W3t = (_Float16*)(ws + off); off += SZ_W3T;
  float*    dlt = (float*)(ws + off);                     // BATCH*120*4 = 15.7 MB

  // 1) weight transpose/convert (tiny; weights then live in L2)
  conv_transpose<<<(HID * KPAD1 + 255) / 256, 256, 0, stream>>>(W1, W1t, 414, HID, KPAD1, HID);
  conv_transpose<<<(HID * HID + 255) / 256, 256, 0, stream>>>(W2, W2t, HID, HID, HID, HID);
  conv_transpose<<<(N3PAD * HID + 255) / 256, 256, 0, stream>>>(W3, W3t, HID, N3VALID, HID, N3PAD);

  // 2) feature build
  build_x<<<BATCH, 128, 0, stream>>>(obs, action, X);

  // 3) MLP via WMMA f16
  gemm_relu_f16<<<dim3(HID / 256, BATCH / 32), 256, 0, stream>>>(X, W1t, b1, H1, KPAD1, KPAD1, KPAD1, HID);
  _Float16* H2 = X;  // X dead after GEMM1
  gemm_relu_f16<<<dim3(HID / 256, BATCH / 32), 256, 0, stream>>>(H1, W2t, b2, H2, HID, HID, HID, HID);
  gemm_delta<<<dim3(1, BATCH / 32), 256, 0, stream>>>(H2, W3t, b3, dstd, dmean, dlt, HID);

  // 4) state integration
  integrate<<<(BATCH * NBODY + 255) / 256, 256, 0, stream>>>(state, dlt, (float*)d_out);

  (void)in_sizes; (void)n_in; (void)out_size; (void)ws_size;
}
